// RNN_arch_2_final_65670049955967
// MI455X (gfx1250) — compile-verified, matrix operand order checked by reading, not run on hardware
//
#include <hip/hip_runtime.h>
#include <hip/hip_bf16.h>
#include <math.h>

typedef __attribute__((ext_vector_type(16))) __bf16 v16bf;
typedef __attribute__((ext_vector_type(8)))  __bf16 v8bf;
typedef __attribute__((ext_vector_type(8)))  float  v8f;

#define TSTEPS 16

// ---- hardware tanh (v_tanh_f32, TRANS pipe) with fallback ----
__device__ __forceinline__ float hw_tanh(float x) {
#if __has_builtin(__builtin_amdgcn_tanhf)
  return __builtin_amdgcn_tanhf(x);
#else
  return tanhf(x);
#endif
}

// ---------------- WMMA fragment helpers ----------------
// A fragment (16x32 bf16, M x K): lane L -> row m = L%16; half = L/16;
// elems 0..7  = K (kb*32 + half*8 + e)
// elems 8..15 = K (kb*32 + 16 + half*8 + (e-8))
__device__ __forceinline__ v16bf wmma_ld_a(const __bf16* buf, int stride,
                                           int kb, int nl, int half) {
  const __bf16* p = buf + nl * stride + kb * 32 + half * 8;
  v8bf lo = *(const v8bf*)p;
  v8bf hi = *(const v8bf*)(p + 16);
  return __builtin_shufflevector(lo, hi, 0,1,2,3,4,5,6,7,8,9,10,11,12,13,14,15);
}

// B fragment: packed in d_ws by the prep kernel so each lane reads 16
// contiguous bf16 (32B) at tile base + lane*32B.
__device__ __forceinline__ v16bf wmma_ld_b(const __bf16* w, int ktiles,
                                           int nt, int kb, int lane) {
  return *(const v16bf*)(w + ((size_t)(nt * ktiles + kb) * 32 + lane) * 16);
}

__device__ __forceinline__ v8f wmma_bf16(v16bf a, v16bf b, v8f c) {
  return __builtin_amdgcn_wmma_f32_16x16x32_bf16(false, a, false, b,
                                                 (short)0, c, false, false);
}

template<int NKB>
__device__ __forceinline__ void preload(v16bf* dst, const __bf16* buf,
                                        int stride, int nl, int half) {
#pragma unroll
  for (int kb = 0; kb < NKB; ++kb) dst[kb] = wmma_ld_a(buf, stride, kb, nl, half);
}

// C tile [16x16]: lane L -> n = L%16, row m = 8*(L/16) + v   (v = VGPR 0..7)

// ---------------- weight pack (prep) ----------------
struct PackDesc { const float* W; int O, I, Kp, Np; unsigned int dst; };
struct PackArgs { PackDesc d[12]; unsigned short* ws; };

__global__ void rnn_pack_kernel(PackArgs pa) {
  PackDesc de = pa.d[blockIdx.y];
  int kt = de.Kp >> 5;
  int total = de.Kp * de.Np;
  int t = blockIdx.x * blockDim.x + threadIdx.x;
  if (t >= total) return;
  int e    = t & 15;
  int lane = (t >> 4) & 31;
  int kb   = (t >> 9) % kt;
  int nt   = (t >> 9) / kt;
  int nl = lane & 15, hf = lane >> 4;
  int kl = (e < 8) ? (hf * 8 + e) : (16 + hf * 8 + (e - 8));
  int k = kb * 32 + kl, n = nt * 16 + nl;
  float v = (k < de.I && n < de.O) ? de.W[(size_t)n * de.I + k] : 0.0f;
  ((__bf16*)pa.ws)[(size_t)de.dst + (unsigned)t] = (__bf16)v;
}

// combined recurrent biases (time-invariant): bsum = bi + bh, fp32, in ws
__global__ void rnn_bias_kernel(const float* b1i, const float* b1h,
                                const float* b2i, const float* b2h,
                                const float* b3i, const float* b3h,
                                float* bsum) {
  int i = threadIdx.x;                   // 256 threads
  bsum[i]       = b1i[i] + b1h[i];
  bsum[256 + i] = b2i[i] + b2h[i];
  bsum[512 + i] = b3i[i] + b3h[i];
}

__global__ void rnn_cue_kernel(const float* cue, float* o_cue, int B) {
  int b = blockIdx.x * blockDim.x + threadIdx.x;
  if (b >= B) return;
  float c = cue[b];
  float c0 = c * 10.0f, c1 = 10.0f * fabsf(c - 1.0f);
  for (int t = 0; t < TSTEPS; ++t) {
    o_cue[((size_t)t * B + b) * 2 + 0] = c0;
    o_cue[((size_t)t * B + b) * 2 + 1] = c1;
  }
}

// ---------------- main persistent RNN kernel ----------------
struct MainArgs {
  const float* x; const float* cue;
  const float* hc1; const float* hc2; const float* hc3; const float* hc4;
  const float* bf1v; const float* b1o;
  const float* bf2v; const float* b2o;
  const float* b3o;  const float* bfc;
  const float* bsum;             // [b1i+b1h | b2i+b2h | b3i+b3h] in ws
  const unsigned short* w[12];   // packed bf16 B-fragments in d_ws
  float* o_oseq; float* o_h1; float* o_h2; float* o_h3; float* o_h4;
  float* o_o1; float* o_o2; float* o_f1; float* o_f2; float* o_ofa;
  int B;
};

// One RNN stack (1 or 2): feed = h4@Wf^T + bfd (stored pre-tanh),
// hnew = tanh(xi@Wi^T + h@Wh^T + (bi+bh) + tanh(feed)), written into hbuf.
__device__ __forceinline__ void run_stack(
    const v16bf* xiA, const v16bf* h4A, const v16bf* hA,
    __bf16* hbuf, const __bf16* wf, const __bf16* wi, const __bf16* wh,
    const float* bfd, const float* bsum,
    float* fout, int lane, int nl, int half) {
  for (int nt = 0; nt < 16; ++nt) {
    v8f fc = {};
#pragma unroll
    for (int kb = 0; kb < 2; ++kb)
      fc = wmma_bf16(h4A[kb], wmma_ld_b(wf, 2, nt, kb, lane), fc);
    float bfv = bfd[nt * 16 + nl];
#pragma unroll
    for (int v = 0; v < 8; ++v) fc[v] += bfv;
#pragma unroll
    for (int v = 0; v < 8; ++v)
      fout[(size_t)(8 * half + v) * 256 + nt * 16 + nl] = fc[v];
    v8f mc = {};
#pragma unroll
    for (int kb = 0; kb < 2; ++kb)
      mc = wmma_bf16(xiA[kb], wmma_ld_b(wi, 2, nt, kb, lane), mc);
#pragma unroll
    for (int kb = 0; kb < 8; ++kb)
      mc = wmma_bf16(hA[kb], wmma_ld_b(wh, 8, nt, kb, lane), mc);
    float bb = bsum[nt * 16 + nl];
#pragma unroll
    for (int v = 0; v < 8; ++v) {
      float hv = hw_tanh(mc[v] + bb + hw_tanh(fc[v]));
      hbuf[(8 * half + v) * 256 + nt * 16 + nl] = (__bf16)hv;
    }
  }
}

__device__ __forceinline__ v8f mm_k256(const v16bf* hA, const __bf16* w,
                                       int nt, int lane) {
  v8f c = {};
#pragma unroll
  for (int kb = 0; kb < 8; ++kb)
    c = wmma_bf16(hA[kb], wmma_ld_b(w, 8, nt, kb, lane), c);
  return c;
}

#define LDS_PER_WAVE 17920   // bf16 elements

__global__ __launch_bounds__(128, 1)
void rnn_main_kernel(MainArgs a) {
  extern __shared__ __bf16 ldsraw[];
  const int lane = threadIdx.x & 31;
  const int wv   = threadIdx.x >> 5;
  const int nl   = lane & 15;
  const int half = lane >> 4;
  const int base = (blockIdx.x * 4 + wv) * 16;
  const int B = a.B;
  if (base >= B) return;             // wave-uniform: EXEC stays all-1s for WMMA

  __bf16* wl   = ldsraw + (size_t)wv * LDS_PER_WAVE;
  __bf16* h1b  = wl;                 // 16x256
  __bf16* h2b  = wl + 4096;          // 16x256
  __bf16* h3b  = wl + 8192;          // 16x256
  __bf16* h4b  = wl + 12288;         // 16x64
  __bf16* midb = wl + 13312;         // 16x160 (cue|out1|out2|zero-pad)
  __bf16* xib  = wl + 15872;         // 16x64
  __bf16* ofb  = wl + 16896;         // 16x64

  const __bf16* wp[12];
#pragma unroll
  for (int i = 0; i < 12; ++i) wp[i] = (const __bf16*)a.w[i];
  const float* bsum1 = a.bsum;
  const float* bsum2 = a.bsum + 256;
  const float* bsum3 = a.bsum + 512;

  // ---- init hidden state (fp32 global -> bf16 LDS) ----
  for (int i = 0; i < 128; ++i) {
    int idx = i * 32 + lane; int r = idx >> 8, c = idx & 255;
    size_t g = (size_t)(base + r) * 256 + c;
    h1b[idx] = (__bf16)a.hc1[g];
    h2b[idx] = (__bf16)a.hc2[g];
    h3b[idx] = (__bf16)a.hc3[g];
  }
  for (int i = 0; i < 32; ++i) {
    int idx = i * 32 + lane; int r = idx >> 6, c = idx & 63;
    h4b[idx] = (__bf16)a.hc4[(size_t)(base + r) * 64 + c];
  }
  if (lane < 16) {
    float cv = a.cue[base + lane];
    midb[lane * 160 + 0] = (__bf16)(cv * 10.0f);
    midb[lane * 160 + 1] = (__bf16)(10.0f * fabsf(cv - 1.0f));
    for (int j = 130; j < 160; ++j) midb[lane * 160 + j] = (__bf16)0.0f;
  }

  for (int t = 0; t < TSTEPS; ++t) {
    // xi -> LDS (bf16)
    const float* xt = a.x + ((size_t)t * B + base) * 64;
    for (int i = 0; i < 32; ++i) {
      int idx = i * 32 + lane;
      xib[idx] = (__bf16)xt[idx];
    }

    v16bf xiA[2], h4A[2], hA[8];
    preload<2>(xiA, xib, 64, nl, half);
    preload<2>(h4A, h4b, 64, nl, half);

    // ---- stack 1 ----
    preload<8>(hA, h1b, 256, nl, half);
    run_stack(xiA, h4A, hA, h1b, wp[2], wp[0], wp[1],
              a.bf1v, bsum1,
              a.o_f1 + ((size_t)t * B + base) * 256, lane, nl, half);
    preload<8>(hA, h1b, 256, nl, half);
    {
      float* o1 = a.o_o1 + ((size_t)t * B + base) * 64;
      for (int nt = 0; nt < 4; ++nt) {
        v8f c = mm_k256(hA, wp[3], nt, lane);
        float bb = a.b1o[nt * 16 + nl];
#pragma unroll
        for (int v = 0; v < 8; ++v) {
          float ov = hw_tanh(c[v] + bb);
          o1[(size_t)(8 * half + v) * 64 + nt * 16 + nl] = ov;
          midb[(8 * half + v) * 160 + 2 + nt * 16 + nl] = (__bf16)ov;
        }
      }
    }

    // ---- stack 2 ----
    preload<8>(hA, h2b, 256, nl, half);
    run_stack(xiA, h4A, hA, h2b, wp[6], wp[4], wp[5],
              a.bf2v, bsum2,
              a.o_f2 + ((size_t)t * B + base) * 256, lane, nl, half);
    preload<8>(hA, h2b, 256, nl, half);
    {
      float* o2 = a.o_o2 + ((size_t)t * B + base) * 64;
      for (int nt = 0; nt < 4; ++nt) {
        v8f c = mm_k256(hA, wp[7], nt, lane);
        float bb = a.b2o[nt * 16 + nl];
#pragma unroll
        for (int v = 0; v < 8; ++v) {
          float ov = hw_tanh(c[v] + bb);
          o2[(size_t)(8 * half + v) * 64 + nt * 16 + nl] = ov;
          midb[(8 * half + v) * 160 + 66 + nt * 16 + nl] = (__bf16)ov;
        }
      }
    }

    // ---- stack 3: h3n = tanh(mid@W3i^T + h3@W3h^T + (b3i+b3h)) ----
    {
      v16bf midA[5];
      preload<5>(midA, midb, 160, nl, half);
      preload<8>(hA, h3b, 256, nl, half);
      for (int nt = 0; nt < 16; ++nt) {
        v8f c = {};
#pragma unroll
        for (int kb = 0; kb < 5; ++kb)
          c = wmma_bf16(midA[kb], wmma_ld_b(wp[8], 5, nt, kb, lane), c);
#pragma unroll
        for (int kb = 0; kb < 8; ++kb)
          c = wmma_bf16(hA[kb], wmma_ld_b(wp[9], 8, nt, kb, lane), c);
        float bb = bsum3[nt * 16 + nl];
#pragma unroll
        for (int v = 0; v < 8; ++v)
          h3b[(8 * half + v) * 256 + nt * 16 + nl] = (__bf16)hw_tanh(c[v] + bb);
      }
    }

    // ---- o = h3n@W3o^T + b3o ; h4 <- o ; ofa = relu(o) ----
    preload<8>(hA, h3b, 256, nl, half);
    {
      float* ofa = a.o_ofa + ((size_t)t * B + base) * 64;
      for (int nt = 0; nt < 4; ++nt) {
        v8f c = mm_k256(hA, wp[10], nt, lane);
        float bb = a.b3o[nt * 16 + nl];
#pragma unroll
        for (int v = 0; v < 8; ++v) {
          float ov = c[v] + bb;
          int ridx = 8 * half + v;
          h4b[ridx * 64 + nt * 16 + nl] = (__bf16)ov;
          float ofv = fmaxf(ov, 0.0f);
          ofa[(size_t)ridx * 64 + nt * 16 + nl] = ofv;
          ofb[ridx * 64 + nt * 16 + nl] = (__bf16)ofv;
          if (t == TSTEPS - 1)
            a.o_h4[(size_t)(base + ridx) * 64 + nt * 16 + nl] = ov;
        }
      }
    }

    // ---- outf = of@Wfc^T + bfc -> oseq[t] (only n<2 valid) ----
    {
      v16bf ofA[2];
      preload<2>(ofA, ofb, 64, nl, half);
      v8f c = {};
#pragma unroll
      for (int kb = 0; kb < 2; ++kb)
        c = wmma_bf16(ofA[kb], wmma_ld_b(wp[11], 2, 0, kb, lane), c);
      if (nl < 2) {
        float bb = a.bfc[nl];
        float* od = a.o_oseq + ((size_t)t * B + base) * 2;
#pragma unroll
        for (int v = 0; v < 8; ++v)
          od[(size_t)(8 * half + v) * 2 + nl] = c[v] + bb;
      }
    }
  }

  // ---- final hidden-state outputs ----
  for (int i = 0; i < 128; ++i) {
    int idx = i * 32 + lane; int r = idx >> 8, c = idx & 255;
    size_t g = (size_t)(base + r) * 256 + c;
    a.o_h1[g] = (float)h1b[idx];
    a.o_h2[g] = (float)h2b[idx];
    a.o_h3[g] = (float)h3b[idx];
  }
}

// ---------------- host launcher ----------------
extern "C" void kernel_launch(void* const* d_in, const int* in_sizes, int n_in,
                              void* d_out, int out_size, void* d_ws, size_t ws_size,
                              hipStream_t stream) {
  (void)n_in; (void)out_size; (void)ws_size;
  const int B = in_sizes[1];          // cue is [B]
  float* out = (float*)d_out;

  // output sub-buffer offsets in return order
  size_t p = 0;
  float* o_oseq = out + p; p += (size_t)TSTEPS * B * 2;
  float* o_h1   = out + p; p += (size_t)B * 256;
  float* o_h2   = out + p; p += (size_t)B * 256;
  float* o_h3   = out + p; p += (size_t)B * 256;
  float* o_h4   = out + p; p += (size_t)B * 64;
  float* o_o1   = out + p; p += (size_t)TSTEPS * B * 64;
  float* o_o2   = out + p; p += (size_t)TSTEPS * B * 64;
  float* o_cue  = out + p; p += (size_t)TSTEPS * B * 2;
  float* o_f1   = out + p; p += (size_t)TSTEPS * B * 256;
  float* o_f2   = out + p; p += (size_t)TSTEPS * B * 256;
  float* o_ofa  = out + p; p += (size_t)TSTEPS * B * 64;

  // pack weights (order: W1i W1h Wf1 W1o W2i W2h Wf2 W2o W3i W3h W3o Wfc)
  static const int widx[12] = {6, 8, 10, 12, 14, 16, 18, 20, 22, 24, 26, 28};
  static const int Os[12]   = {256, 256, 256, 64, 256, 256, 256, 64, 256, 256, 64, 2};
  static const int Is[12]   = {64, 256, 64, 256, 64, 256, 64, 256, 130, 256, 256, 64};
  PackArgs pa;
  unsigned int off = 0;
  const unsigned short* wptr[12];
  for (int i = 0; i < 12; ++i) {
    int Kp = ((Is[i] + 31) / 32) * 32;
    int Np = ((Os[i] + 15) / 16) * 16;
    pa.d[i].W = (const float*)d_in[widx[i]];
    pa.d[i].O = Os[i]; pa.d[i].I = Is[i];
    pa.d[i].Kp = Kp;   pa.d[i].Np = Np;
    pa.d[i].dst = off;
    wptr[i] = (const unsigned short*)d_ws + off;
    off += (unsigned)Kp * Np;                 // total 353,280 bf16 = ~706 KB of ws
  }
  pa.ws = (unsigned short*)d_ws;
  dim3 pgrid((65536 + 255) / 256, 12);
  rnn_pack_kernel<<<pgrid, 256, 0, stream>>>(pa);

  // combined biases, fp32, right after packed weights (offset is 4B-aligned)
  float* bsum = (float*)((char*)d_ws + (size_t)off * sizeof(unsigned short));
  rnn_bias_kernel<<<1, 256, 0, stream>>>(
      (const float*)d_in[7],  (const float*)d_in[9],
      (const float*)d_in[15], (const float*)d_in[17],
      (const float*)d_in[23], (const float*)d_in[25], bsum);

  rnn_cue_kernel<<<(B + 255) / 256, 256, 0, stream>>>((const float*)d_in[1], o_cue, B);

  MainArgs ma;
  ma.x = (const float*)d_in[0];  ma.cue = (const float*)d_in[1];
  ma.hc1 = (const float*)d_in[2]; ma.hc2 = (const float*)d_in[3];
  ma.hc3 = (const float*)d_in[4]; ma.hc4 = (const float*)d_in[5];
  ma.bf1v = (const float*)d_in[11]; ma.b1o = (const float*)d_in[13];
  ma.bf2v = (const float*)d_in[19]; ma.b2o = (const float*)d_in[21];
  ma.b3o = (const float*)d_in[27];  ma.bfc = (const float*)d_in[29];
  ma.bsum = bsum;
  for (int i = 0; i < 12; ++i) ma.w[i] = wptr[i];
  ma.o_oseq = o_oseq; ma.o_h1 = o_h1; ma.o_h2 = o_h2; ma.o_h3 = o_h3;
  ma.o_h4 = o_h4; ma.o_o1 = o_o1; ma.o_o2 = o_o2;
  ma.o_f1 = o_f1; ma.o_f2 = o_f2; ma.o_ofa = o_ofa;
  ma.B = B;

  int nblocks = (B + 63) / 64;                 // 64 rows (4 waves x 16) per block
  size_t shmem = (size_t)4 * LDS_PER_WAVE * sizeof(unsigned short);  // 143,360 B
  rnn_main_kernel<<<nblocks, 128, shmem, stream>>>(ma);
}
